// DGCNN_38800734552538
// MI455X (gfx1250) — compile-verified, matrix operand order checked by reading, not run on hardware
//
#include <hip/hip_runtime.h>
#include <hip/hip_bf16.h>

// ---------------------------------------------------------------------------
// DGCNN head for MI455X (gfx1250, wave32, WMMA + async-to-LDS staging).
// ---------------------------------------------------------------------------

typedef __attribute__((ext_vector_type(16))) _Float16 v16h;
typedef __attribute__((ext_vector_type(8)))  float    v8f;

#define BATCH 4
#define NPTS  4096
#define K_NN  20
#define OUTK  8
#define MPTS  1024          // NPTS/4 after FPS
#define OUTD  256
#define CHUNK 1024          // rows of the Gram matrix computed per launch

// ---- WMMA fragment loader: 16 rows (or cols) of a row-major [*,64] f16 array
// CDNA5 16-bit A/B layout: lane l holds row (l&15); half=(l>>4) selects K
// sub-blocks {half*8..half*8+7} and {16+half*8..16+half*8+7} -> two 16B loads.
// Works for both global and LDS-backed pointers (addrspace inferred after
// inlining; LDS case lowers to ds_load_b128).
__device__ __forceinline__ v16h frag_ld64(const _Float16* base, int row, int koff, int lane) {
    int r    = row + (lane & 15);
    int half = lane >> 4;
    const _Float16* p = base + (size_t)r * 64 + koff + half * 8;
    union { v16h v; float4 q[2]; } u;
    u.q[0] = *reinterpret_cast<const float4*>(p);
    u.q[1] = *reinterpret_cast<const float4*>(p + 16);
    return u.v;
}

__device__ __forceinline__ v8f wmma_f16(v16h a, v16h b, v8f c) {
    return __builtin_amdgcn_wmma_f32_16x16x32_f16(false, a, false, b,
                                                  (short)0, c, false, false);
}

// ---- CDNA5 async global->LDS copy (ASYNCcnt-tracked) -----------------------
__device__ __forceinline__ void async_ld_b128(unsigned lds_off, const void* gptr) {
    asm volatile("global_load_async_to_lds_b128 %0, %1, off"
                 :: "v"(lds_off), "v"((unsigned long long)(size_t)gptr)
                 : "memory");
}
__device__ __forceinline__ void wait_async0() {
    asm volatile("s_wait_asynccnt 0x0" ::: "memory");
}

// ---- Pairwise -||xi-xj||^2 via WMMA: S = 2*X1 X1^T - xx_i - xx_j -----------
// 4-wave workgroup computes a 64x64 tile; A/B panels (8KB each) staged into
// LDS with global_load_async_to_lds_b128; each wave does 8 WMMAs (16x64 strip).
__global__ void __launch_bounds__(128)
pairdist_wmma(const _Float16* __restrict__ X1h,
              const float* __restrict__ xx,
              float* __restrict__ S,
              int b, int row_base) {
    __shared__ __align__(16) _Float16 smem[8192];   // [0:4096)=A panel, [4096:8192)=B panel
    int lane = threadIdx.x & 31;
    int wid  = threadIdx.x >> 5;
    int j_block = blockIdx.x * 64;
    int i_block = row_base + blockIdx.y * 64;
    const _Float16* Xb  = X1h + (size_t)b * NPTS * 64;
    const float*    xxb = xx  + (size_t)b * NPTS;

    // stage panels: 2 x 8KB = 2 x 512 chunks of 16B, 128 threads x 4 chunks
    const char* gA = (const char*)(Xb + (size_t)i_block * 64);
    const char* gB = (const char*)(Xb + (size_t)j_block * 64);
    unsigned sbase = (unsigned)(size_t)smem;
    #pragma unroll
    for (int i = 0; i < 4; ++i) {
        int c = i * 128 + threadIdx.x;              // 0..511
        async_ld_b128(sbase + c * 16,        gA + c * 16);
        async_ld_b128(sbase + 8192 + c * 16, gB + c * 16);
    }
    wait_async0();
    __syncthreads();

    // A fragments for this wave's 16 rows (from LDS)
    v16h a0 = frag_ld64(smem, wid * 16, 0,  lane);
    v16h a1 = frag_ld64(smem, wid * 16, 32, lane);

    v8f acc[4];
    #pragma unroll
    for (int jt = 0; jt < 4; ++jt) {
        v16h b0 = frag_ld64(smem + 4096, jt * 16, 0,  lane);
        v16h b1 = frag_ld64(smem + 4096, jt * 16, 32, lane);
        v8f c = {};
        c = wmma_f16(a0, b0, c);
        c = wmma_f16(a1, b1, c);
        acc[jt] = c;
    }

    int nn = lane & 15, mb = (lane >> 4) * 8;
    int i_wave = i_block + wid * 16;
    float xr[8];
    #pragma unroll
    for (int r = 0; r < 8; ++r) xr[r] = xxb[i_wave + mb + r];
    #pragma unroll
    for (int jt = 0; jt < 4; ++jt) {
        int n = j_block + jt * 16 + nn;
        float xxj = xxb[n];
        #pragma unroll
        for (int r = 0; r < 8; ++r) {
            int m = i_wave + mb + r;
            S[(size_t)(m - row_base) * NPTS + n] = 2.0f * acc[jt][r] - xr[r] - xxj;
        }
    }
}

// ---- C[M,O] = A[M,64] * W[O,64]^T via WMMA; each wave -> 16x64 strip -------
__global__ void gemm64_wmma(const _Float16* __restrict__ A,
                            const _Float16* __restrict__ W,
                            float* __restrict__ C, int Ocols) {
    int lane = threadIdx.x & 31;
    int o0 = blockIdx.x * 64;
    int m0 = blockIdx.y * 16;
    v16h a0 = frag_ld64(A, m0, 0,  lane);
    v16h a1 = frag_ld64(A, m0, 32, lane);
    v8f acc[4];
    #pragma unroll
    for (int t = 0; t < 4; ++t) {
        v16h b0 = frag_ld64(W, o0 + t * 16, 0,  lane);
        v16h b1 = frag_ld64(W, o0 + t * 16, 32, lane);
        v8f c = {};
        c = wmma_f16(a0, b0, c);
        c = wmma_f16(a1, b1, c);
        acc[t] = c;
    }
    int nn = lane & 15, mb = (lane >> 4) * 8;
    #pragma unroll
    for (int t = 0; t < 4; ++t)
        #pragma unroll
        for (int r = 0; r < 8; ++r)
            C[(size_t)(m0 + mb + r) * Ocols + o0 + t * 16 + nn] = acc[t][r];
}

// ---- Brute-force KNN in 3D with LDS tiling ---------------------------------
template <int KK>
__global__ void knn_bruteforce(const float* __restrict__ q,
                               const float* __restrict__ s,
                               int* __restrict__ idx, int Nq, int Ns) {
    __shared__ float sp[256 * 3];
    int b  = blockIdx.y;
    int qi = blockIdx.x * 256 + threadIdx.x;
    bool active = qi < Nq;
    float qx = 0.f, qy = 0.f, qz = 0.f;
    if (active) {
        const float* qp = q + ((size_t)b * Nq + qi) * 3;
        qx = qp[0]; qy = qp[1]; qz = qp[2];
    }
    float tv[KK]; int ti[KK];
    #pragma unroll
    for (int k = 0; k < KK; ++k) { tv[k] = -1e30f; ti[k] = 0; }
    for (int t0 = 0; t0 < Ns; t0 += 256) {
        __syncthreads();
        int li = t0 + threadIdx.x;
        if (li < Ns) {
            const float* pp = s + ((size_t)b * Ns + li) * 3;
            sp[threadIdx.x * 3 + 0] = pp[0];
            sp[threadIdx.x * 3 + 1] = pp[1];
            sp[threadIdx.x * 3 + 2] = pp[2];
        }
        __syncthreads();
        if (active) {
            int tile = (Ns - t0) < 256 ? (Ns - t0) : 256;
            for (int j = 0; j < tile; ++j) {
                float dx = qx - sp[j * 3 + 0];
                float dy = qy - sp[j * 3 + 1];
                float dz = qz - sp[j * 3 + 2];
                float nd = -(dx * dx + dy * dy + dz * dz);
                if (nd > tv[KK - 1]) {
                    int p = KK - 1;
                    while (p > 0 && tv[p - 1] < nd) {
                        tv[p] = tv[p - 1]; ti[p] = ti[p - 1]; --p;
                    }
                    tv[p] = nd; ti[p] = t0 + j;
                }
            }
        }
    }
    if (active)
        for (int k = 0; k < KK; ++k)
            idx[((size_t)b * Nq + qi) * KK + k] = ti[k];
}

// ---- Top-20 per row of the precomputed neg-distance chunk ------------------
__global__ void topk_rows(const float* __restrict__ S, int* __restrict__ idx,
                          int rows, int b, int row_base) {
    int r = blockIdx.x * blockDim.x + threadIdx.x;
    if (r >= rows) return;
    const float* row = S + (size_t)r * NPTS;
    float tv[K_NN]; int ti[K_NN];
    #pragma unroll
    for (int k = 0; k < K_NN; ++k) { tv[k] = -1e30f; ti[k] = 0; }
    for (int j = 0; j < NPTS; ++j) {
        float v = row[j];
        if (v > tv[K_NN - 1]) {
            int p = K_NN - 1;
            while (p > 0 && tv[p - 1] < v) {
                tv[p] = tv[p - 1]; ti[p] = ti[p - 1]; --p;
            }
            tv[p] = v; ti[p] = j;
        }
    }
    int n = row_base + r;
    for (int k = 0; k < K_NN; ++k)
        idx[((size_t)b * NPTS + n) * K_NN + k] = ti[k];
}

// ---- Edge-conv1 factorization: P = x*w1a^T, Q = x*(w1b-w1a)^T --------------
__global__ void pq1_kernel(const float* __restrict__ x,
                           const float* __restrict__ w1,
                           float* __restrict__ P, float* __restrict__ Q) {
    int t = blockIdx.x;          // global point index b*N+n
    int o = threadIdx.x;         // channel 0..63
    __shared__ float sx[3];
    if (o < 3) sx[o] = x[(size_t)t * 3 + o];
    __syncthreads();
    float a0 = w1[o * 6 + 0], a1 = w1[o * 6 + 1], a2 = w1[o * 6 + 2];
    float b0 = w1[o * 6 + 3], b1 = w1[o * 6 + 4], b2 = w1[o * 6 + 5];
    P[(size_t)t * 64 + o] = sx[0] * a0 + sx[1] * a1 + sx[2] * a2;
    Q[(size_t)t * 64 + o] = sx[0] * (b0 - a0) + sx[1] * (b1 - a1) + sx[2] * (b2 - a2);
}

// ---- Batch-norm statistics over y[b,n,k,o] = P[j]+Q[n] ---------------------
__global__ void stats_edge(const float* __restrict__ P, const float* __restrict__ Q,
                           const int* __restrict__ idx, float* __restrict__ sums) {
    int o = threadIdx.x;         // 64 channels
    float s = 0.f, ss = 0.f;
    int base = blockIdx.x * 64;
    for (int u = 0; u < 64; ++u) {
        int t = base + u;
        int b = t / NPTS;
        float qv = Q[(size_t)t * 64 + o];
        for (int k = 0; k < K_NN; ++k) {
            int j = idx[(size_t)t * K_NN + k];
            float y = P[((size_t)b * NPTS + j) * 64 + o] + qv;
            s += y; ss += y * y;
        }
    }
    atomicAdd(&sums[o], s);
    atomicAdd(&sums[64 + o], ss);
}

__global__ void finalize_bn(const float* __restrict__ sums,
                            const float* __restrict__ g, const float* __restrict__ be,
                            float* __restrict__ scale, float* __restrict__ shift,
                            int C, float invcnt) {
    int o = blockIdx.x * blockDim.x + threadIdx.x;
    if (o >= C) return;
    float m = sums[o] * invcnt;
    float v = sums[C + o] * invcnt - m * m;
    float sc = g[o] * rsqrtf(v + 1e-5f);
    scale[o] = sc;
    shift[o] = be[o] - m * sc;
}

// ---- BN + LeakyReLU(0.2) + max over k --------------------------------------
__global__ void edge_max(const float* __restrict__ P, const float* __restrict__ Q,
                         const int* __restrict__ idx,
                         const float* __restrict__ scale, const float* __restrict__ shift,
                         float* __restrict__ Xout, _Float16* __restrict__ Xouth) {
    int t = blockIdx.x;          // b*N+n
    int o = threadIdx.x;         // 64
    int b = t / NPTS;
    float sc = scale[o], sh = shift[o];
    float qv = Q[(size_t)t * 64 + o];
    float m = -1e30f;
    for (int k = 0; k < K_NN; ++k) {
        int j = idx[(size_t)t * K_NN + k];
        float y = sc * (P[((size_t)b * NPTS + j) * 64 + o] + qv) + sh;
        y = y > 0.f ? y : 0.2f * y;
        m = fmaxf(m, y);
    }
    Xout[(size_t)t * 64 + o] = m;
    if (Xouth) Xouth[(size_t)t * 64 + o] = (_Float16)m;
}

__global__ void rowsumsq(const float* __restrict__ X, float* __restrict__ xx, int total) {
    int t = blockIdx.x * blockDim.x + threadIdx.x;
    if (t >= total) return;
    float s = 0.f;
    for (int c = 0; c < 64; ++c) { float v = X[(size_t)t * 64 + c]; s += v * v; }
    xx[t] = s;
}

// split w2[64,128] into Wa=w2[:, :64] and Wd=w2[:,64:]-w2[:, :64] (f16)
__global__ void prep_w2(const float* __restrict__ w2,
                        _Float16* __restrict__ Wa, _Float16* __restrict__ Wd) {
    int t = blockIdx.x * blockDim.x + threadIdx.x;
    if (t >= 64 * 64) return;
    int o = t >> 6, k = t & 63;
    float a = w2[o * 128 + k], b = w2[o * 128 + 64 + k];
    Wa[t] = (_Float16)a;
    Wd[t] = (_Float16)(b - a);
}

__global__ void to_half(const float* __restrict__ src, _Float16* __restrict__ dst, int n) {
    int t = blockIdx.x * blockDim.x + threadIdx.x;
    if (t < n) dst[t] = (_Float16)src[t];
}

// ---- Furthest point sampling: one block per batch, serial argmax loop ------
__global__ void fps_kernel(const float* __restrict__ x, int* __restrict__ idx2) {
    __shared__ float dist[NPTS];
    __shared__ float rv[256];
    __shared__ int   ri[256];
    __shared__ float lp[3];
    __shared__ int   s_last;
    int b = blockIdx.x, tid = threadIdx.x;
    for (int i = tid; i < NPTS; i += 256) dist[i] = 1e10f;
    if (tid == 0) { idx2[b * MPTS] = 0; s_last = 0; }
    __syncthreads();
    for (int it = 1; it < MPTS; ++it) {
        if (tid == 0) {
            const float* p = x + ((size_t)b * NPTS + s_last) * 3;
            lp[0] = p[0]; lp[1] = p[1]; lp[2] = p[2];
        }
        __syncthreads();
        float bv = -1.f; int bi = 0;
        for (int i = tid; i < NPTS; i += 256) {
            const float* p = x + ((size_t)b * NPTS + i) * 3;
            float dx = p[0] - lp[0], dy = p[1] - lp[1], dz = p[2] - lp[2];
            float d = dx * dx + dy * dy + dz * dz;
            float nd = fminf(dist[i], d);
            dist[i] = nd;
            if (nd > bv) { bv = nd; bi = i; }
        }
        rv[tid] = bv; ri[tid] = bi;
        __syncthreads();
        for (int s = 128; s > 0; s >>= 1) {
            if (tid < s && rv[tid + s] > rv[tid]) {
                rv[tid] = rv[tid + s]; ri[tid] = ri[tid + s];
            }
            __syncthreads();
        }
        if (tid == 0) { s_last = ri[0]; idx2[b * MPTS + it] = ri[0]; }
        __syncthreads();
    }
}

__global__ void gather_p2(const float* __restrict__ x, const int* __restrict__ idx2,
                          float* __restrict__ p2) {
    int t = blockIdx.x * blockDim.x + threadIdx.x;
    if (t >= BATCH * MPTS) return;
    int b = t / MPTS;
    int j = idx2[t];
    const float* p = x + ((size_t)b * NPTS + j) * 3;
    p2[(size_t)t * 3 + 0] = p[0];
    p2[(size_t)t * 3 + 1] = p[1];
    p2[(size_t)t * 3 + 2] = p[2];
}

__global__ void agg_kernel(const float* __restrict__ X2, const int* __restrict__ idxo,
                           float* __restrict__ agg, _Float16* __restrict__ aggh) {
    int t = blockIdx.x;          // b*M+m
    int o = threadIdx.x;         // 64
    int b = t / MPTS;
    float s = 0.f;
    for (int k = 0; k < OUTK; ++k) {
        int j = idxo[(size_t)t * OUTK + k];
        s += X2[((size_t)b * NPTS + j) * 64 + o];
    }
    s *= (1.0f / OUTK);
    agg[(size_t)t * 64 + o] = s;
    aggh[(size_t)t * 64 + o] = (_Float16)s;
}

__global__ void stats_out(const float* __restrict__ Y, float* __restrict__ sums) {
    int o = threadIdx.x;         // 256 channels
    float s = 0.f, ss = 0.f;
    int r0 = blockIdx.x * 64;
    for (int u = 0; u < 64; ++u) {
        float y = Y[(size_t)(r0 + u) * OUTD + o];
        s += y; ss += y * y;
    }
    atomicAdd(&sums[o], s);
    atomicAdd(&sums[OUTD + o], ss);
}

__global__ void final_out(const float* __restrict__ Y,
                          const float* __restrict__ scale, const float* __restrict__ shift,
                          float* __restrict__ out) {
    int g = blockIdx.x * blockDim.x + threadIdx.x;     // over B*OUTD*M
    if (g >= BATCH * OUTD * MPTS) return;
    int b = g / (OUTD * MPTS);
    int rem = g % (OUTD * MPTS);
    int o = rem / MPTS, m = rem % MPTS;
    float y = Y[((size_t)b * MPTS + m) * OUTD + o];
    out[g] = scale[o] * y + shift[o];
}

__global__ void write_tail(const float* __restrict__ p2, float* __restrict__ out) {
    int t = blockIdx.x * blockDim.x + threadIdx.x;
    const int tot = BATCH * MPTS * 3;
    const int tm = BATCH * MPTS;
    if (t < tot) { out[t] = p2[t]; out[tot + t] = p2[t]; }
    if (t < tm)  { out[2 * tot + t] = 1.0f; out[2 * tot + tm + t] = 1.0f; }
}

// ---------------------------------------------------------------------------
static inline size_t rup(size_t x) { return (x + 255) & ~(size_t)255; }

extern "C" void kernel_launch(void* const* d_in, const int* in_sizes, int n_in,
                              void* d_out, int out_size, void* d_ws, size_t ws_size,
                              hipStream_t stream) {
    const float* x  = (const float*)d_in[0];
    const float* w1 = (const float*)d_in[1];
    const float* g1 = (const float*)d_in[2];
    const float* b1 = (const float*)d_in[3];
    const float* w2 = (const float*)d_in[4];
    const float* g2 = (const float*)d_in[5];
    const float* b2 = (const float*)d_in[6];
    const float* wf = (const float*)d_in[7];
    const float* gf = (const float*)d_in[8];
    const float* bf = (const float*)d_in[9];
    const float* wc = (const float*)d_in[10];
    const float* gc = (const float*)d_in[11];
    const float* bc = (const float*)d_in[12];
    (void)in_sizes; (void)n_in; (void)out_size; (void)ws_size;

    const int BN = BATCH * NPTS;       // 16384
    const int BM = BATCH * MPTS;       // 4096

    // ---- workspace layout
    char* w = (char*)d_ws;
    size_t off = 0;
    auto alloc = [&](size_t bytes) { char* p = w + off; off += rup(bytes); return p; };
    float*    Schunk = (float*)   alloc((size_t)CHUNK * NPTS * 4);   // 16 MB
    float*    X1     = (float*)   alloc((size_t)BN * 64 * 4);
    _Float16* X1h    = (_Float16*)alloc((size_t)BN * 64 * 2);
    float*    X2     = (float*)   alloc((size_t)BN * 64 * 4);
    int*      idx1   = (int*)     alloc((size_t)BN * K_NN * 4);
    int*      idxA   = (int*)     alloc((size_t)BN * K_NN * 4);
    float*    P      = (float*)   alloc((size_t)BN * 64 * 4);
    float*    Q      = (float*)   alloc((size_t)BN * 64 * 4);
    _Float16* Wa_h   = (_Float16*)alloc(64 * 64 * 2);
    _Float16* Wd_h   = (_Float16*)alloc(64 * 64 * 2);
    _Float16* wf_h   = (_Float16*)alloc(OUTD * 64 * 2);
    _Float16* wc_h   = (_Float16*)alloc(OUTD * 64 * 2);
    float*    xx     = (float*)   alloc((size_t)BN * 4);
    int*      idx2   = (int*)     alloc((size_t)BM * 4);
    float*    p2     = (float*)   alloc((size_t)BM * 3 * 4);
    int*      idxo   = (int*)     alloc((size_t)BM * OUTK * 4);
    float*    agg    = (float*)   alloc((size_t)BM * 64 * 4);
    _Float16* aggh   = (_Float16*)alloc((size_t)BM * 64 * 2);
    float*    Yf     = (float*)   alloc((size_t)BM * OUTD * 4);
    float*    Yc     = (float*)   alloc((size_t)BM * OUTD * 4);
    float*    stats  = (float*)   alloc(2 * OUTD * 4);
    float*    scale  = (float*)   alloc(OUTD * 4);
    float*    shift  = (float*)   alloc(OUTD * 4);

    float* outF = (float*)d_out;
    float* outC = outF + (size_t)BATCH * OUTD * MPTS;
    float* outT = outC + (size_t)BATCH * OUTD * MPTS;

    // ---- edge-conv block 1 (KNN on 3D coords, factored conv) ----
    knn_bruteforce<K_NN><<<dim3(NPTS / 256, BATCH), 256, 0, stream>>>(x, x, idx1, NPTS, NPTS);
    pq1_kernel<<<BN, 64, 0, stream>>>(x, w1, P, Q);
    hipMemsetAsync(stats, 0, 2 * 64 * 4, stream);
    stats_edge<<<BN / 64, 64, 0, stream>>>(P, Q, idx1, stats);
    finalize_bn<<<1, 64, 0, stream>>>(stats, g1, b1, scale, shift, 64,
                                      1.0f / ((float)BN * K_NN));
    edge_max<<<BN, 64, 0, stream>>>(P, Q, idx1, scale, shift, X1, X1h);

    // ---- edge-conv block 2: WMMA Gram matrix KNN + WMMA factored conv ----
    rowsumsq<<<(BN + 127) / 128, 128, 0, stream>>>(X1, xx, BN);
    prep_w2<<<(64 * 64 + 127) / 128, 128, 0, stream>>>(w2, Wa_h, Wd_h);
    gemm64_wmma<<<dim3(1, BN / 16), 32, 0, stream>>>(X1h, Wa_h, P, 64);
    gemm64_wmma<<<dim3(1, BN / 16), 32, 0, stream>>>(X1h, Wd_h, Q, 64);
    for (int b = 0; b < BATCH; ++b)
        for (int rb = 0; rb < NPTS / CHUNK; ++rb) {
            int row_base = rb * CHUNK;
            pairdist_wmma<<<dim3(NPTS / 64, CHUNK / 64), 128, 0, stream>>>(
                X1h, xx, Schunk, b, row_base);
            topk_rows<<<CHUNK / 128, 128, 0, stream>>>(Schunk, idxA, CHUNK, b, row_base);
        }
    hipMemsetAsync(stats, 0, 2 * 64 * 4, stream);
    stats_edge<<<BN / 64, 64, 0, stream>>>(P, Q, idxA, stats);
    finalize_bn<<<1, 64, 0, stream>>>(stats, g2, b2, scale, shift, 64,
                                      1.0f / ((float)BN * K_NN));
    edge_max<<<BN, 64, 0, stream>>>(P, Q, idxA, scale, shift, X2, (_Float16*)nullptr);

    // ---- FPS + output KNN aggregation ----
    fps_kernel<<<BATCH, 256, 0, stream>>>(x, idx2);
    gather_p2<<<(BM + 127) / 128, 128, 0, stream>>>(x, idx2, p2);
    knn_bruteforce<OUTK><<<dim3((MPTS + 255) / 256, BATCH), 256, 0, stream>>>(
        p2, x, idxo, MPTS, NPTS);
    agg_kernel<<<BM, 64, 0, stream>>>(X2, idxo, agg, aggh);

    // ---- output projections (WMMA) + BN + transposed store ----
    to_half<<<(OUTD * 64 + 255) / 256, 256, 0, stream>>>(wf, wf_h, OUTD * 64);
    to_half<<<(OUTD * 64 + 255) / 256, 256, 0, stream>>>(wc, wc_h, OUTD * 64);
    gemm64_wmma<<<dim3(OUTD / 64, BM / 16), 32, 0, stream>>>(aggh, wf_h, Yf, OUTD);
    gemm64_wmma<<<dim3(OUTD / 64, BM / 16), 32, 0, stream>>>(aggh, wc_h, Yc, OUTD);

    hipMemsetAsync(stats, 0, 2 * OUTD * 4, stream);
    stats_out<<<BM / 64, OUTD, 0, stream>>>(Yf, stats);
    finalize_bn<<<1, OUTD, 0, stream>>>(stats, gf, bf, scale, shift, OUTD, 1.0f / (float)BM);
    final_out<<<(BATCH * OUTD * MPTS + 255) / 256, 256, 0, stream>>>(Yf, scale, shift, outF);

    hipMemsetAsync(stats, 0, 2 * OUTD * 4, stream);
    stats_out<<<BM / 64, OUTD, 0, stream>>>(Yc, stats);
    finalize_bn<<<1, OUTD, 0, stream>>>(stats, gc, bc, scale, shift, OUTD, 1.0f / (float)BM);
    final_out<<<(BATCH * OUTD * MPTS + 255) / 256, 256, 0, stream>>>(Yc, scale, shift, outC);

    write_tail<<<(BATCH * MPTS * 3 + 255) / 256, 256, 0, stream>>>(p2, outT);
}
